// RNNLayer_49538152792627
// MI455X (gfx1250) — compile-verified, matrix operand order checked by reading, not run on hardware
//
#include <hip/hip_runtime.h>

typedef _Float16 f16;
typedef __attribute__((ext_vector_type(16))) _Float16 v16h;
typedef __attribute__((ext_vector_type(8)))  _Float16 v8h;
typedef __attribute__((ext_vector_type(4)))  _Float16 v4h;
typedef __attribute__((ext_vector_type(8)))  float    v8f;
typedef __attribute__((ext_vector_type(4)))  float    v4f;
typedef int v4i_vs __attribute__((vector_size(16)));

#define NHID 768
#define HDIM 128
#define NDIR 12
#define SEQ  512
#define BATCH 16
#define CAT  1536
#define DFF  3072
#define MROWS 8192   // BATCH*SEQ

// ---------------------------------------------------------------------------
// CDNA5 async global->LDS staging (ASYNCcnt path), guarded so compile can't
// break if the toolchain lacks the builtins; falls back to ld+ds_store.
// Builtin signature (from hipcc diagnostic): param0 = v4i AS(1)* (global src).
// ---------------------------------------------------------------------------
#if defined(__has_builtin)
#if __has_builtin(__builtin_amdgcn_global_load_async_to_lds_b128) && \
    __has_builtin(__builtin_amdgcn_s_wait_asynccnt)
#define USE_ASYNC 1
#endif
#endif
#ifndef USE_ASYNC
#define USE_ASYNC 0
#endif

__device__ __forceinline__ void cp16_to_lds(const f16* g, f16* l) {
#if USE_ASYNC
  __builtin_amdgcn_global_load_async_to_lds_b128(
      (__attribute__((address_space(1))) v4i_vs*)(uintptr_t)g,
      (__attribute__((address_space(3))) v4i_vs*)(unsigned)(uintptr_t)l, 0, 0);
#else
  *(v8h*)l = *(const v8h*)g;
#endif
}

// ---------------------------------------------------------------------------
// WMMA operand loader: row-major f16 region.
// Per ISA: lane l holds row (l&15); K elements 0..7 at k = (l>>4)*8,
// elements 8..15 at k = 16 + (l>>4)*8.  -> two 16B chunk loads.
// ---------------------------------------------------------------------------
union OpU { v16h v; v8h h[2]; };
__device__ __forceinline__ v16h ldop(const f16* p) {
  OpU o;
  o.h[0] = *(const v8h*)(p);
  o.h[1] = *(const v8h*)(p + 16);
  return o.v;
}

__device__ __forceinline__ v8f wmma32(v16h a, v16h b, v8f c) {
  return __builtin_amdgcn_wmma_f32_16x16x32_f16(false, a, false, b, (short)0, c,
                                                false, false);
}

// ---------------------------------------------------------------------------
// f32 -> f16 conversion (grid-stride)
// ---------------------------------------------------------------------------
__global__ void cvt_f16_kernel(const float* __restrict__ src,
                               f16* __restrict__ dst, long n) {
  long i = (long)blockIdx.x * blockDim.x + threadIdx.x;
  long stride = (long)gridDim.x * blockDim.x;
  for (; i < n; i += stride) dst[i] = (f16)src[i];
}

__global__ void bias_sum_kernel(const float* __restrict__ a,
                                const float* __restrict__ b,
                                float* __restrict__ o, int n) {
  int i = blockIdx.x * blockDim.x + threadIdx.x;
  if (i < n) o[i] = a[i] + b[i];
}

// ---------------------------------------------------------------------------
// Unified WMMA GEMM:  C[M,N] = A[M,K] * B[N,K]^T + bias[col]
// block tile 128x128, 8 waves (4 row x 2 col), wave tile 32x64 (2x4 wmma).
// Double-buffered LDS, async global->LDS staging when available.
// epi==0: outF[k*strideOutK + row*N + col] = acc + bias
// epi==1: outH[row*N + col] = (f16) gelu(acc + bias)    (exact erf GELU)
// ---------------------------------------------------------------------------
#define LDST 56  // padded LDS row stride (f16): 112B rows, 16B aligned, conflict-free

__global__ __launch_bounds__(256) void gemm_wmma_kernel(
    const f16* __restrict__ A, const f16* __restrict__ B,
    const float* __restrict__ bias, float* __restrict__ outF,
    f16* __restrict__ outH, int M, int N, int K, long strideBk,
    long strideBiasK, long strideOutK, int epi) {
  __shared__ f16 sA[2][128 * LDST];
  __shared__ f16 sB[2][128 * LDST];

  const int kz = blockIdx.z;
  const f16* Bk = B + (long)kz * strideBk;
  const float* biask = bias + (long)kz * strideBiasK;

  const int tid = threadIdx.x;
  const int lane = tid & 31;
  const int wave = tid >> 5;
  const int wm = wave & 3;   // 4 wave rows (32 each)
  const int wn = wave >> 2;  // 2 wave cols (64 each)
  const int rowBase = blockIdx.y * 128;
  const int colBase = blockIdx.x * 128;
  const int mn = lane & 15;
  const int kh8 = (lane >> 4) * 8;

  v8f acc[2][4];
#pragma unroll
  for (int i = 0; i < 2; ++i)
#pragma unroll
    for (int j = 0; j < 4; ++j)
#pragma unroll
      for (int r = 0; r < 8; ++r) acc[i][j][r] = 0.f;

  // stage one 128x32 A tile + 128x32 B tile into buffer p (4 chunks/thread)
  auto stage = [&](int p, int kk) {
#pragma unroll
    for (int it = 0; it < 2; ++it) {
      int slot = tid + it * 256;
      int r = slot >> 2;
      int c = (slot & 3) * 8;
      cp16_to_lds(A + (long)(rowBase + r) * K + kk + c, &sA[p][r * LDST + c]);
      cp16_to_lds(Bk + (long)(colBase + r) * K + kk + c, &sB[p][r * LDST + c]);
    }
  };

  stage(0, 0);
  int cur = 0;
  for (int kk = 0; kk < K; kk += 32) {
    if (kk + 32 < K) {
      stage(cur ^ 1, kk + 32);
#if USE_ASYNC
      __builtin_amdgcn_s_wait_asynccnt(4);  // older group (4 instrs) done
#endif
    } else {
#if USE_ASYNC
      __builtin_amdgcn_s_wait_asynccnt(0);
#endif
    }
    __syncthreads();

    v16h a[2], b[4];
#pragma unroll
    for (int i = 0; i < 2; ++i)
      a[i] = ldop(&sA[cur][(wm * 32 + i * 16 + mn) * LDST + kh8]);
#pragma unroll
    for (int j = 0; j < 4; ++j)
      b[j] = ldop(&sB[cur][(wn * 64 + j * 16 + mn) * LDST + kh8]);

#pragma unroll
    for (int i = 0; i < 2; ++i)
#pragma unroll
      for (int j = 0; j < 4; ++j) acc[i][j] = wmma32(a[i], b[j], acc[i][j]);

    __syncthreads();
    cur ^= 1;
  }

  // epilogue: D element r of lane l -> row = 8*(l>>4)+r, col = l&15 (per tile)
  const int half8 = (lane >> 4) * 8;
#pragma unroll
  for (int i = 0; i < 2; ++i) {
#pragma unroll
    for (int j = 0; j < 4; ++j) {
      int col = colBase + wn * 64 + j * 16 + mn;
      float bv = biask[col];
#pragma unroll
      for (int r = 0; r < 8; ++r) {
        int row = rowBase + wm * 32 + i * 16 + half8 + r;
        float v = acc[i][j][r] + bv;
        if (epi == 0) {
          outF[(long)kz * strideOutK + (long)row * N + col] = v;
        } else {
          float g = 0.5f * v * (1.f + erff(v * 0.70710678118654752f));
          outH[(long)row * N + col] = (f16)g;
        }
      }
    }
  }
}

// ---------------------------------------------------------------------------
// Recurrent multi-BiLSTM kernel: 12 blocks (one per direction) x 512 threads
// (16 waves). W_hh B-operands live in registers (wave owns gate cols
// [32w,32w+32)); h lives in LDS as the shared A-operand; z and c in LDS.
// Per step: z = xp[t] + h*Whh^T via 8 WMMAs/wave -> gates -> h,c update ->
// h written to LDS (f16) + cat output (f16, reference layout).
// ---------------------------------------------------------------------------
#define SH_STRIDE 136  // 16 x 136 f16, 272B rows -> conflict free A reads
#define SZ_STRIDE 516  // 16 x 516 f32
#define SC_STRIDE 132  // 16 x 132 f32

__global__ __launch_bounds__(512) void bilstm_kernel(
    const f16* __restrict__ Whh, const float* __restrict__ xp,
    f16* __restrict__ cat) {
  __shared__ f16 sh[16 * SH_STRIDE];
  __shared__ float sz[16 * SZ_STRIDE];
  __shared__ float sc[16 * SC_STRIDE];

  const int k = blockIdx.x;
  const int tid = threadIdx.x;
  const int lane = tid & 31;
  const int wave = tid >> 5;  // 0..15
  const int mn = lane & 15;
  const int kh8 = (lane >> 4) * 8;
  const int half8 = kh8;

  // preload W_hh operands: wave covers z cols [wave*32, wave*32+32)
  v16h Bop[2][4];
  const f16* WhhK = Whh + (long)k * 512 * HDIM;
#pragma unroll
  for (int j = 0; j < 2; ++j) {
    int g = wave * 32 + j * 16 + mn;  // W_hh row (gate column of z)
#pragma unroll
    for (int ks = 0; ks < 4; ++ks)
      Bop[j][ks] = ldop(WhhK + (long)g * HDIM + ks * 32 + kh8);
  }

  // zero init h (f16) and c (f32)
  for (int i = tid; i < 16 * SH_STRIDE; i += 512) sh[i] = (f16)0.f;
  for (int i = tid; i < 16 * SC_STRIDE; i += 512) sc[i] = 0.f;
  __syncthreads();

  const long xpK = (long)k * BATCH * SEQ * 512;
  const int catBase = (k < 6) ? (k * 256) : ((k - 6) * 256 + 128);
  const int gm = tid & 15;         // gate-stage batch row
  const int gc0 = (tid >> 4) * 4;  // gate-stage h-col base (4 cols)

  for (int s = 0; s < SEQ; ++s) {
    const int t = (k < 6) ? s : (SEQ - 1 - s);

    // ---- WMMA stage: z = xp[t] + h * Whh^T ----
    v16h Aop[4];
#pragma unroll
    for (int ks = 0; ks < 4; ++ks)
      Aop[ks] = ldop(sh + mn * SH_STRIDE + ks * 32 + kh8);

#pragma unroll
    for (int j = 0; j < 2; ++j) {
      const int col = wave * 32 + j * 16 + mn;
      v8f z;
#pragma unroll
      for (int r = 0; r < 8; ++r) {
        int b = half8 + r;
        z[r] = xp[xpK + ((long)b * SEQ + t) * 512 + col];
      }
#pragma unroll
      for (int ks = 0; ks < 4; ++ks) z = wmma32(Aop[ks], Bop[j][ks], z);
#pragma unroll
      for (int r = 0; r < 8; ++r) sz[(half8 + r) * SZ_STRIDE + col] = z[r];
    }
    __syncthreads();

    // ---- gate stage: 512 threads x 4 h-columns ----
    v4f iv = *(const v4f*)(sz + gm * SZ_STRIDE + gc0);
    v4f fv = *(const v4f*)(sz + gm * SZ_STRIDE + 128 + gc0);
    v4f gv = *(const v4f*)(sz + gm * SZ_STRIDE + 256 + gc0);
    v4f ov = *(const v4f*)(sz + gm * SZ_STRIDE + 384 + gc0);
    v4f cc = *(const v4f*)(sc + gm * SC_STRIDE + gc0);
    v4h hh;
#pragma unroll
    for (int q = 0; q < 4; ++q) {
      float si = 1.f / (1.f + expf(-iv[q]));
      float sf = 1.f / (1.f + expf(-fv[q]));
      float so = 1.f / (1.f + expf(-ov[q]));
      float c2 = sf * cc[q] + si * tanhf(gv[q]);
      cc[q] = c2;
      float h = so * tanhf(c2);
      hh[q] = (f16)h;
      sh[gm * SH_STRIDE + gc0 + q] = (f16)h;
    }
    *(v4f*)(sc + gm * SC_STRIDE + gc0) = cc;
    // cat[b][t][head*256 + dir*128 + col]  (backward written time-reversed)
    *(v4h*)(cat + ((long)gm * SEQ + t) * CAT + catBase + gc0) = hh;
    __syncthreads();
  }
}

// ---------------------------------------------------------------------------
// residual + LayerNorm: one block per row (768 = 3*256 cols)
// ---------------------------------------------------------------------------
__global__ __launch_bounds__(256) void resid_ln_kernel(
    const float* __restrict__ emb, const float* __restrict__ proj,
    const float* __restrict__ gamma, const float* __restrict__ beta,
    float* __restrict__ out) {
  const long row = blockIdx.x;
  const int tid = threadIdx.x;
  const int lane = tid & 31;
  const int wave = tid >> 5;
  __shared__ float redS[8];
  __shared__ float redQ[8];

  float v[3];
  float sum = 0.f, sq = 0.f;
#pragma unroll
  for (int j = 0; j < 3; ++j) {
    int c = tid + j * 256;
    float x = emb[row * NHID + c] + proj[row * NHID + c];
    v[j] = x;
    sum += x;
    sq += x * x;
  }
#pragma unroll
  for (int o = 16; o; o >>= 1) {
    sum += __shfl_xor(sum, o, 32);
    sq += __shfl_xor(sq, o, 32);
  }
  if (lane == 0) { redS[wave] = sum; redQ[wave] = sq; }
  __syncthreads();
  float ts = 0.f, tq = 0.f;
#pragma unroll
  for (int i = 0; i < 8; ++i) { ts += redS[i]; tq += redQ[i]; }
  const float mu = ts * (1.f / NHID);
  const float var = tq * (1.f / NHID) - mu * mu;
  const float rs = rsqrtf(var + 1e-5f);
#pragma unroll
  for (int j = 0; j < 3; ++j) {
    int c = tid + j * 256;
    out[row * NHID + c] = (v[j] - mu) * rs * gamma[c] + beta[c];
  }
}

// ---------------------------------------------------------------------------
// host launcher
// ---------------------------------------------------------------------------
extern "C" void kernel_launch(void* const* d_in, const int* in_sizes, int n_in,
                              void* d_out, int out_size, void* d_ws,
                              size_t ws_size, hipStream_t stream) {
  (void)in_sizes; (void)n_in; (void)out_size; (void)ws_size;
  const float* emb   = (const float*)d_in[0];
  const float* W_ih  = (const float*)d_in[1];
  const float* W_hh  = (const float*)d_in[2];
  const float* b_ih  = (const float*)d_in[3];
  const float* b_hh  = (const float*)d_in[4];
  const float* W_int = (const float*)d_in[5];
  const float* b_int = (const float*)d_in[6];
  const float* W_out = (const float*)d_in[7];
  const float* b_out = (const float*)d_in[8];
  const float* gamma = (const float*)d_in[9];
  const float* beta  = (const float*)d_in[10];
  float* out = (float*)d_out;

  char* ws = (char*)d_ws;
  size_t off = 0;
  auto alloc = [&](size_t bytes) -> void* {
    void* p = ws + off;
    off += (bytes + 255) & ~(size_t)255;
    return p;
  };

  const long nX   = (long)MROWS * NHID;
  const long nWih = (long)NDIR * 512 * NHID;
  const long nWhh = (long)NDIR * 512 * HDIM;
  const long nWin = (long)DFF * CAT;
  const long nWou = (long)NHID * DFF;

  f16*   xh    = (f16*)alloc(nX * 2);
  f16*   Wih_h = (f16*)alloc(nWih * 2);
  f16*   Whh_h = (f16*)alloc(nWhh * 2);
  f16*   Win_h = (f16*)alloc(nWin * 2);
  f16*   Wou_h = (f16*)alloc(nWou * 2);
  float* bias1 = (float*)alloc((size_t)NDIR * 512 * 4);
  float* xp    = (float*)alloc((size_t)NDIR * BATCH * SEQ * 512 * 4);  // 201MB
  f16*   cat_h = (f16*)alloc((size_t)MROWS * CAT * 2);
  f16*   int_h = (f16*)alloc((size_t)MROWS * DFF * 2);
  float* proj  = (float*)alloc((size_t)MROWS * NHID * 4);

  // 1) precision staging
  cvt_f16_kernel<<<2048, 256, 0, stream>>>(emb, xh, nX);
  cvt_f16_kernel<<<2048, 256, 0, stream>>>(W_ih, Wih_h, nWih);
  cvt_f16_kernel<<<1024, 256, 0, stream>>>(W_hh, Whh_h, nWhh);
  cvt_f16_kernel<<<2048, 256, 0, stream>>>(W_int, Win_h, nWin);
  cvt_f16_kernel<<<2048, 256, 0, stream>>>(W_out, Wou_h, nWou);
  bias_sum_kernel<<<(NDIR * 512 + 255) / 256, 256, 0, stream>>>(
      b_ih, b_hh, bias1, NDIR * 512);

  // 2) input projection: xp[k] = x * W_ih[k]^T + (b_ih+b_hh)[k]
  {
    dim3 grid(512 / 128, MROWS / 128, NDIR);
    gemm_wmma_kernel<<<grid, 256, 0, stream>>>(
        xh, Wih_h, bias1, xp, nullptr, MROWS, 512, NHID,
        (long)512 * NHID, (long)512, (long)MROWS * 512, /*epi=*/0);
  }

  // 3) recurrent scan (12 directions in parallel)
  bilstm_kernel<<<NDIR, 512, 0, stream>>>(Whh_h, xp, cat_h);

  // 4) FFN1: inter = gelu(cat * W_int^T + b_int)  -> f16
  {
    dim3 grid(DFF / 128, MROWS / 128, 1);
    gemm_wmma_kernel<<<grid, 256, 0, stream>>>(
        cat_h, Win_h, b_int, nullptr, int_h, MROWS, DFF, CAT, 0L, 0L, 0L,
        /*epi=*/1);
  }

  // 5) FFN2: proj = inter * W_out^T + b_out  -> f32
  {
    dim3 grid(NHID / 128, MROWS / 128, 1);
    gemm_wmma_kernel<<<grid, 256, 0, stream>>>(
        int_h, Wou_h, b_out, proj, nullptr, MROWS, NHID, DFF, 0L, 0L, 0L,
        /*epi=*/0);
  }

  // 6) residual + LayerNorm
  resid_ln_kernel<<<MROWS, 256, 0, stream>>>(emb, proj, gamma, beta, out);
}
